// WeightGenerator_39178691674622
// MI455X (gfx1250) — matrix operand order repaired and checked by reading, not compiled, not run on hardware
//
#include <hip/hip_runtime.h>

// ---------------------------------------------------------------------------
// CDNA5 (gfx1250) implementation. wave32. WMMA bf16 16x16x32 for the 64->64
// 3x3 conv layers (implicit GEMM, K = 9*64 = 576), VALU for the rest.
// Layer weights are DMA'd into LDS by the Tensor Data Mover (TENSORcnt) and
// fed to WMMA via ds reads.
// ---------------------------------------------------------------------------

typedef __attribute__((ext_vector_type(16))) __bf16 v16bf;
typedef __attribute__((ext_vector_type(8)))  float  v8f;
typedef __attribute__((ext_vector_type(4)))  int    v4i;
typedef __attribute__((ext_vector_type(8)))  int    v8i;

union AFrag { v16bf v; unsigned int u[8]; };

// ---------------------------------------------------------------------------
// Workspace layout (bytes)
// ---------------------------------------------------------------------------
#define OFF_WF_I1 0ULL
#define OFF_WF_I2 73728ULL
#define OFF_WF_B1 147456ULL
#define OFF_WF_B2 221184ULL
#define OFF_IIN   294912ULL                     // 32*64*64*64 bf16 = 16 MiB
#define OFF_I1    (OFF_IIN + 16777216ULL)
#define OFF_I2    (OFF_I1  + 16777216ULL)
#define OFF_IW    (OFF_I2  + 16777216ULL)       // 32*64*64 f32
#define OFF_BGIN  (OFF_IW  + 524288ULL)         // 4*128*128*64 bf16
#define OFF_BG1   (OFF_BGIN + 8388608ULL)
#define OFF_BG2   (OFF_BG1  + 8388608ULL)
#define OFF_BGW   (OFF_BG2  + 8388608ULL)       // 4*128*128 f32

#define WFRAG_ELEMS  36864                      // 18*4*32*16 bf16 per layer
#define WFRAG_BYTES  73728
#define WFRAG_DWORDS 18432

// ---------------------------------------------------------------------------
// Repack (3,3,64,64) f32 weights into WMMA B-fragment order, bf16.
// Layout: [kstep 0..17][ntile 0..3][lane 0..31][elem 0..15]
// value(ks,nt,lane,e) = W[k = ks*32 + 16*(e/8) + 8*(lane/16) + e%8][n = nt*16 + lane%16]
// K ordering: k = tap*64 + cin  (tap = 3*dy+dx)
// ---------------------------------------------------------------------------
__global__ __launch_bounds__(256) void repack_w_bf16(const float* __restrict__ W,
                                                     __bf16* __restrict__ out)
{
  const int i    = blockIdx.x * 256 + threadIdx.x;   // exactly 36864 threads
  const int e    =  i        & 15;
  const int lane = (i >> 4)  & 31;
  const int nt   = (i >> 9)  &  3;
  const int ks   =  i >> 11;                         // 0..17
  const int klocal = ((e & 8) << 1) | ((lane >> 4) << 3) | (e & 7);
  const int k   = ks * 32 + klocal;
  const int tap = k >> 6;
  const int ci  = k & 63;
  const int n   = (nt << 4) | (lane & 15);
  out[i] = (__bf16)W[(tap * 64 + ci) * 64 + n];
}

// (B,h,w,C,MI) f32 -> (B*MI,h,w,C) bf16
__global__ __launch_bounds__(256) void pack_instance_bf16(const float* __restrict__ in,
                                                          __bf16* __restrict__ out)
{
  const int i  = blockIdx.x * 256 + threadIdx.x;     // 8388608 threads
  const int c  =  i        & 63;
  const int x  = (i >> 6)  & 63;
  const int y  = (i >> 12) & 63;
  const int mi = (i >> 18) &  7;
  const int b  =  i >> 21;
  out[i] = (__bf16)in[(((b * 64 + y) * 64 + x) * 64 + c) * 8 + mi];
}

__global__ __launch_bounds__(256) void pack_bg_bf16(const float* __restrict__ in,
                                                    __bf16* __restrict__ out)
{
  const int i = blockIdx.x * 256 + threadIdx.x;      // 4194304 threads
  out[i] = (__bf16)in[i];
}

// ---------------------------------------------------------------------------
// TDM: DMA the layer's 73728-byte weight-fragment buffer into LDS.
// Tensor DMA Descriptor built per CDNA5 ISA sec 8.3 / 8.4:
//   group0: count=1 | lds_addr | global_addr[56:0] | type=2
//   group1: data_size=4B, tensor = one row of 18432 dwords, tile = same row
// Issued by one wave; completion via s_wait_tensorcnt then workgroup barrier.
// ---------------------------------------------------------------------------
__device__ static inline void tdm_load_weights_to_lds(const void* gsrc, unsigned lds_off)
{
  union { v4i v; unsigned w[4]; } g0;
  union { v8i v; unsigned w[8]; } g1;
  const unsigned long long ga = (unsigned long long)gsrc;
  g0.w[0] = 1u;                                         // count=1 (valid user D#)
  g0.w[1] = lds_off;                                    // lds_addr (bytes)
  g0.w[2] = (unsigned)(ga & 0xffffffffu);               // global_addr[31:0]
  g0.w[3] = (unsigned)((ga >> 32) & 0x01ffffffu)        // global_addr[56:32]
          | (2u << 30);                                 // type = 2 ("image")
  g1.w[0] = 2u << 16;                                   // data_size = 4 bytes
  g1.w[1] = (WFRAG_DWORDS & 0xffffu) << 16;             // tensor_dim0[15:0]
  g1.w[2] = (WFRAG_DWORDS >> 16) | (1u << 16);          // tensor_dim0[31:16], tensor_dim1=1
  g1.w[3] = (WFRAG_DWORDS & 0xffffu) << 16;             // tile_dim0 = 18432
  g1.w[4] = 1u;                                         // tile_dim1 = 1
  g1.w[5] = WFRAG_DWORDS;                               // tensor_dim0_stride[31:0]
  g1.w[6] = (WFRAG_DWORDS & 0xffffu) << 16;             // tensor_dim1_stride[15:0]
  g1.w[7] = (WFRAG_DWORDS >> 16);                       // tensor_dim1_stride[47:16]
  asm volatile("tensor_load_to_lds %0, %1" :: "s"(g0.v), "s"(g1.v) : "memory");
}

// ---------------------------------------------------------------------------
// 3x3 SAME conv, 64 -> 64 channels, + bias + ReLU, bf16 in/out, f32 accum.
// One wave32 per 32-pixel x 64-channel output tile (2 M-tiles x 4 N-tiles =
// 8 WMMA accumulators). Implicit GEMM, K = 576 in 18 steps of 32 (one 3x3
// tap + 32-channel slice per step).
// Weight B-fragments for the whole layer (72 KB) are staged into LDS by the
// TDM once per block and read back with ds loads (reused by 8 waves).
// A fragment per ISA "16-bit A 16x32" layout: lane row m = lane%16,
//   pair i (reg i): k_local = (i<4 ? 2i : 8+2i) + 8*(lane/16)
// ---------------------------------------------------------------------------
__global__ __launch_bounds__(256) void conv3x3_relu_wmma_bf16(
    const __bf16* __restrict__ in, const __bf16* __restrict__ wfrag,
    const float* __restrict__ bias, __bf16* __restrict__ out,
    int H, int W, int logW, int logHW)
{
  extern __shared__ __bf16 smem[];               // WFRAG_ELEMS bf16 = 72 KB

  // ---- TDM-stage the layer's weight fragments into LDS (wave 0 issues) ----
  if (threadIdx.x < 32) {
    tdm_load_weights_to_lds(wfrag, (unsigned)(unsigned long long)(uintptr_t)smem);
    __builtin_amdgcn_s_wait_tensorcnt(0);
  }
  __syncthreads();

  const int lane = threadIdx.x & 31;
  const int g    = lane >> 4;
  const int tile = blockIdx.x * 8 + (threadIdx.x >> 5);   // 32-pixel tile id

  // two 16-pixel M-tiles per wave
  const int p0 = tile * 32 + (lane & 15);
  const int p1 = p0 + 16;
  const int nimg0 = p0 >> logHW, y0 = (p0 >> logW) & (H - 1), x0 = p0 & (W - 1);
  const int nimg1 = p1 >> logHW, y1 = (p1 >> logW) & (H - 1), x1 = p1 & (W - 1);

  // warm L2/WGP$ for this wave's input rows (global_prefetch_b8)
  __builtin_prefetch(in + (((nimg0 * H + y0) * W + x0) << 6), 0, 3);
  __builtin_prefetch(in + (((nimg1 * H + y1) * W + x1) << 6), 0, 3);

  v8f acc[2][4] = {};

  for (int ks = 0; ks < 18; ++ks) {
    const int tap   = ks >> 1;
    const int cbase = (ks & 1) << 5;
    const int dy    = tap / 3 - 1;
    const int dx    = tap % 3 - 1;

    AFrag a0, a1;
    {
      const int yy = y0 + dy, xx = x0 + dx;
      if (yy >= 0 && yy < H && xx >= 0 && xx < W) {
        const unsigned int* s =
            (const unsigned int*)(in + (((nimg0 * H + yy) * W + xx) << 6) + cbase + (g << 3));
        a0.u[0] = s[0];  a0.u[1] = s[1];  a0.u[2] = s[2];  a0.u[3] = s[3];
        a0.u[4] = s[8];  a0.u[5] = s[9];  a0.u[6] = s[10]; a0.u[7] = s[11];
      } else {
#pragma unroll
        for (int j = 0; j < 8; ++j) a0.u[j] = 0u;
      }
    }
    {
      const int yy = y1 + dy, xx = x1 + dx;
      if (yy >= 0 && yy < H && xx >= 0 && xx < W) {
        const unsigned int* s =
            (const unsigned int*)(in + (((nimg1 * H + yy) * W + xx) << 6) + cbase + (g << 3));
        a1.u[0] = s[0];  a1.u[1] = s[1];  a1.u[2] = s[2];  a1.u[3] = s[3];
        a1.u[4] = s[8];  a1.u[5] = s[9];  a1.u[6] = s[10]; a1.u[7] = s[11];
      } else {
#pragma unroll
        for (int j = 0; j < 8; ++j) a1.u[j] = 0u;
      }
    }

    // load ALL four B fragments first -> one ds wait covers the whole K-step
    const __bf16* wb = smem + (((ks * 4) * 32 + lane) << 4);
    v16bf bf[4];
#pragma unroll
    for (int t = 0; t < 4; ++t) bf[t] = *(const v16bf*)(wb + t * 512);

#pragma unroll
    for (int t = 0; t < 4; ++t) {
      acc[0][t] = __builtin_amdgcn_wmma_f32_16x16x32_bf16(
          false, a0.v, false, bf[t], (short)0, acc[0][t], false, false);
      acc[1][t] = __builtin_amdgcn_wmma_f32_16x16x32_bf16(
          false, a1.v, false, bf[t], (short)0, acc[1][t], false, false);
    }
  }

  // C/D layout: VGPR r: lanes 0-15 -> M=r, N=lane; lanes 16-31 -> M=8+r, N=lane-16
  const int nn = lane & 15;
#pragma unroll
  for (int mt = 0; mt < 2; ++mt) {
#pragma unroll
    for (int t = 0; t < 4; ++t) {
      const int ch = (t << 4) | nn;
      const float bv = bias[ch];
#pragma unroll
      for (int r = 0; r < 8; ++r) {
        const int q = tile * 32 + mt * 16 + (g << 3) + r;
        float v = acc[mt][t][r] + bv;
        v = v > 0.f ? v : 0.f;
        out[(q << 6) + ch] = (__bf16)v;
      }
    }
  }
}

// ---------------------------------------------------------------------------
// 3x3 SAME conv, 64 -> 1 channel, + bias + ReLU. bf16 in, f32 out. VALU.
// ---------------------------------------------------------------------------
__global__ __launch_bounds__(256) void conv3x3_relu_c64to1(
    const __bf16* __restrict__ in, const float* __restrict__ W9,
    const float* __restrict__ b, float* __restrict__ out,
    int H, int W, int logW, int logHW)
{
  const int p    = blockIdx.x * 256 + threadIdx.x;
  const int nimg = p >> logHW;
  const int y    = (p >> logW) & (H - 1);
  const int x    =  p          & (W - 1);
  float acc = 0.f;
  for (int tap = 0; tap < 9; ++tap) {
    const int yy = y + tap / 3 - 1;
    const int xx = x + tap % 3 - 1;
    if (yy < 0 || yy >= H || xx < 0 || xx >= W) continue;
    const __bf16* s  = in + (((nimg * H + yy) * W + xx) << 6);
    const float*  wv = W9 + tap * 64;
#pragma unroll 8
    for (int c = 0; c < 64; ++c) acc += (float)s[c] * wv[c];
  }
  const float v = acc + b[0];
  out[p] = v > 0.f ? v : 0.f;
}

// ---------------------------------------------------------------------------
// Compose: per output pixel (block), 64 channels (threads). Implements
// _resize_and_pad semantics: trunc-to-int sizes/fronts, max(size,1),
// clip src to [0,n_in-1], zero outside box (so exp(rw)=1 outside box),
// mask resampled over the full canvas (always valid).
// ---------------------------------------------------------------------------
__global__ __launch_bounds__(64) void compose_softmax_blend(
    const float* __restrict__ inst,   // (4,64,64,64,8)
    const float* __restrict__ bg,     // (4,128,128,64)
    const float* __restrict__ bbox,   // (4,4,8)
    const float* __restrict__ mask,   // (4,56,56,8)
    const float* __restrict__ iw,     // (32,64,64)
    const float* __restrict__ bgw,    // (4,128,128)
    float* __restrict__ out)          // (4,128,128,64)
{
  const int pix = blockIdx.x;          // 65536 pixels
  const int b = pix >> 14, y = (pix >> 7) & 127, x = pix & 127;
  const int c = threadIdx.x;

  // mask sampling: full box, n_in = 56, size = 128
  const float msy = fminf(fmaxf((y + 0.5f) * 0.4375f - 0.5f, 0.f), 55.f);
  const int   my0 = (int)msy;
  const int   my1 = min(my0 + 1, 55);
  const float mfy = msy - (float)my0;
  const float msx = fminf(fmaxf((x + 0.5f) * 0.4375f - 0.5f, 0.f), 55.f);
  const int   mx0 = (int)msx;
  const int   mx1 = min(mx0 + 1, 55);
  const float mfx = msx - (float)mx0;

  float ew[8];
  const float ewbg = expf(bgw[pix]);
  float denom = ewbg;

#pragma unroll
  for (int mi = 0; mi < 8; ++mi) {
    const float y1 = bbox[(b * 4 + 0) * 8 + mi], y2 = bbox[(b * 4 + 1) * 8 + mi];
    const float x1 = bbox[(b * 4 + 2) * 8 + mi], x2 = bbox[(b * 4 + 3) * 8 + mi];
    const int ys = max((int)(128.f * (y2 - y1)), 1);
    const int xs = max((int)(128.f * (x2 - x1)), 1);
    const int ly = y - (int)(128.f * y1);
    const int lx = x - (int)(128.f * x1);
    float rw = 0.f;
    if (ly >= 0 && ly < ys && lx >= 0 && lx < xs) {
      const float sy  = fminf(fmaxf(((float)ly + 0.5f) * (64.f / (float)ys) - 0.5f, 0.f), 63.f);
      const int   iy0 = (int)sy;
      const int   iy1 = min(iy0 + 1, 63);
      const float fy  = sy - (float)iy0;
      const float sx  = fminf(fmaxf(((float)lx + 0.5f) * (64.f / (float)xs) - 0.5f, 0.f), 63.f);
      const int   ix0 = (int)sx;
      const int   ix1 = min(ix0 + 1, 63);
      const float fx  = sx - (float)ix0;
      const float* ip = iw + (((b << 3) + mi) << 12);
      const float v00 = ip[(iy0 << 6) + ix0], v01 = ip[(iy0 << 6) + ix1];
      const float v10 = ip[(iy1 << 6) + ix0], v11 = ip[(iy1 << 6) + ix1];
      rw = v00 * (1.f - fy) * (1.f - fx) + v01 * (1.f - fy) * fx
         + v10 * fy * (1.f - fx)         + v11 * fy * fx;
    }
    const float* mp = mask + b * (56 * 56 * 8) + mi;
    const float m00 = mp[(my0 * 56 + mx0) * 8], m01 = mp[(my0 * 56 + mx1) * 8];
    const float m10 = mp[(my1 * 56 + mx0) * 8], m11 = mp[(my1 * 56 + mx1) * 8];
    const float rm = m00 * (1.f - mfy) * (1.f - mfx) + m01 * (1.f - mfy) * mfx
                   + m10 * mfy * (1.f - mfx)         + m11 * mfy * mfx;
    const float e = expf(rw) * rm;
    ew[mi] = e;
    denom += e;
  }

  const float inv = 1.f / denom;
  float acc = ewbg * inv * bg[(pix << 6) + c];

#pragma unroll
  for (int mi = 0; mi < 8; ++mi) {
    const float y1 = bbox[(b * 4 + 0) * 8 + mi], y2 = bbox[(b * 4 + 1) * 8 + mi];
    const float x1 = bbox[(b * 4 + 2) * 8 + mi], x2 = bbox[(b * 4 + 3) * 8 + mi];
    const int ys = max((int)(128.f * (y2 - y1)), 1);
    const int xs = max((int)(128.f * (x2 - x1)), 1);
    const int ly = y - (int)(128.f * y1);
    const int lx = x - (int)(128.f * x1);
    if (ly >= 0 && ly < ys && lx >= 0 && lx < xs) {
      const float sy  = fminf(fmaxf(((float)ly + 0.5f) * (64.f / (float)ys) - 0.5f, 0.f), 63.f);
      const int   iy0 = (int)sy;
      const int   iy1 = min(iy0 + 1, 63);
      const float fy  = sy - (float)iy0;
      const float sx  = fminf(fmaxf(((float)lx + 0.5f) * (64.f / (float)xs) - 0.5f, 0.f), 63.f);
      const int   ix0 = (int)sx;
      const int   ix1 = min(ix0 + 1, 63);
      const float fx  = sx - (float)ix0;
      // inst index: (((b*64+yy)*64+xx)*64 + c)*8 + mi
      const int base = (c << 3) + mi + (b << 21);
      const float s00 = inst[base + ((iy0 << 12) + (ix0 << 6)) * 8];
      const float s01 = inst[base + ((iy0 << 12) + (ix1 << 6)) * 8];
      const float s10 = inst[base + ((iy1 << 12) + (ix0 << 6)) * 8];
      const float s11 = inst[base + ((iy1 << 12) + (ix1 << 6)) * 8];
      const float s = s00 * (1.f - fy) * (1.f - fx) + s01 * (1.f - fy) * fx
                    + s10 * fy * (1.f - fx)         + s11 * fy * fx;
      acc += s * ew[mi] * inv;
    }
  }
  out[(pix << 6) + c] = acc;
}

// ---------------------------------------------------------------------------
extern "C" void kernel_launch(void* const* d_in, const int* in_sizes, int n_in,
                              void* d_out, int out_size, void* d_ws, size_t ws_size,
                              hipStream_t stream)
{
  (void)in_sizes; (void)n_in; (void)out_size; (void)ws_size;

  const float* inst = (const float*)d_in[0];
  const float* bg   = (const float*)d_in[1];
  const float* bbox = (const float*)d_in[2];
  const float* mask = (const float*)d_in[3];
  const float* iW1  = (const float*)d_in[4];
  const float* ib1  = (const float*)d_in[5];
  const float* iW2  = (const float*)d_in[6];
  const float* ib2  = (const float*)d_in[7];
  const float* iW3  = (const float*)d_in[8];
  const float* ib3  = (const float*)d_in[9];
  const float* bW1  = (const float*)d_in[10];
  const float* bb1  = (const float*)d_in[11];
  const float* bW2  = (const float*)d_in[12];
  const float* bb2  = (const float*)d_in[13];
  const float* bW3  = (const float*)d_in[14];
  const float* bb3  = (const float*)d_in[15];

  char* ws = (char*)d_ws;
  __bf16* wfI1 = (__bf16*)(ws + OFF_WF_I1);
  __bf16* wfI2 = (__bf16*)(ws + OFF_WF_I2);
  __bf16* wfB1 = (__bf16*)(ws + OFF_WF_B1);
  __bf16* wfB2 = (__bf16*)(ws + OFF_WF_B2);
  __bf16* iin  = (__bf16*)(ws + OFF_IIN);
  __bf16* i1b  = (__bf16*)(ws + OFF_I1);
  __bf16* i2b  = (__bf16*)(ws + OFF_I2);
  float*  iwf  = (float*) (ws + OFF_IW);
  __bf16* bgin = (__bf16*)(ws + OFF_BGIN);
  __bf16* bg1  = (__bf16*)(ws + OFF_BG1);
  __bf16* bg2  = (__bf16*)(ws + OFF_BG2);
  float*  bgwf = (float*) (ws + OFF_BGW);

  // weight repack + input packing
  repack_w_bf16<<<144, 256, 0, stream>>>(iW1, wfI1);
  repack_w_bf16<<<144, 256, 0, stream>>>(iW2, wfI2);
  repack_w_bf16<<<144, 256, 0, stream>>>(bW1, wfB1);
  repack_w_bf16<<<144, 256, 0, stream>>>(bW2, wfB2);
  pack_instance_bf16<<<32768, 256, 0, stream>>>(inst, iin);
  pack_bg_bf16<<<16384, 256, 0, stream>>>(bg, bgin);

  // instance conv stack: 32 images 64x64 -> 131072 px -> 4096 tiles32 -> 512 blocks
  conv3x3_relu_wmma_bf16<<<512, 256, WFRAG_BYTES, stream>>>(iin, wfI1, ib1, i1b, 64, 64, 6, 12);
  conv3x3_relu_wmma_bf16<<<512, 256, WFRAG_BYTES, stream>>>(i1b, wfI2, ib2, i2b, 64, 64, 6, 12);
  conv3x3_relu_c64to1<<<512, 256, 0, stream>>>(i2b, iW3, ib3, iwf, 64, 64, 6, 12);

  // bg conv stack: 4 images 128x128 -> 65536 px -> 2048 tiles32 -> 256 blocks
  conv3x3_relu_wmma_bf16<<<256, 256, WFRAG_BYTES, stream>>>(bgin, wfB1, bb1, bg1, 128, 128, 7, 14);
  conv3x3_relu_wmma_bf16<<<256, 256, WFRAG_BYTES, stream>>>(bg1, wfB2, bb2, bg2, 128, 128, 7, 14);
  conv3x3_relu_c64to1<<<256, 256, 0, stream>>>(bg2, bW3, bb3, bgwf, 128, 128, 7, 14);

  // final blend
  compose_softmax_blend<<<65536, 64, 0, stream>>>(inst, bg, bbox, mask, iwf, bgwf,
                                                  (float*)d_out);
}
